// EGAT_71468255805605
// MI455X (gfx1250) — matrix operand order repaired and checked by reading, not compiled, not run on hardware
//
#include <hip/hip_runtime.h>

// Problem constants (match reference). N and E are multiples of 16.
#define NN 50000
#define EE 800000

typedef __attribute__((ext_vector_type(16))) __bf16 v16bf;
typedef __attribute__((ext_vector_type(8)))  __bf16 v8bf;
typedef __attribute__((ext_vector_type(8)))  float  v8f;
typedef __attribute__((ext_vector_type(4)))  unsigned u32x4;
typedef __attribute__((ext_vector_type(8)))  int      i32x8;
typedef __attribute__((ext_vector_type(4)))  int      i32x4;

// Pack two f32 into one dword of bf16 (round-to-nearest-even).
__device__ __forceinline__ unsigned pack2bf(float lo, float hi) {
  unsigned a = __builtin_bit_cast(unsigned, lo);
  unsigned b = __builtin_bit_cast(unsigned, hi);
  a += 0x7FFFu + ((a >> 16) & 1u);
  b += 0x7FFFu + ((b >> 16) & 1u);
  return (a >> 16) | (b & 0xFFFF0000u);
}

// f32 -> bf16 streaming convert (4 elements / thread).
__global__ __launch_bounds__(256) void cvt_bf16(
    const float4* __restrict__ in, uint2* __restrict__ out, size_t n4) {
  size_t i = (size_t)blockIdx.x * blockDim.x + threadIdx.x;
  if (i >= n4) return;
  float4 v = in[i];
  out[i] = make_uint2(pack2bf(v.x, v.y), pack2bf(v.z, v.w));
}

// PReLU + f32 -> bf16 convert (layer-0 outputs feed only layer-1 GEMMs).
__global__ __launch_bounds__(256) void prelu_cvt_bf16(
    const float4* __restrict__ in, const float* __restrict__ a,
    uint2* __restrict__ out, size_t n4) {
  size_t i = (size_t)blockIdx.x * blockDim.x + threadIdx.x;
  if (i >= n4) return;
  float aa = *a;
  float4 v = in[i];
  float x = v.x >= 0.f ? v.x : aa * v.x;
  float y = v.y >= 0.f ? v.y : aa * v.y;
  float z = v.z >= 0.f ? v.z : aa * v.z;
  float w = v.w >= 0.f ? v.w : aa * v.w;
  out[i] = make_uint2(pack2bf(x, y), pack2bf(z, w));
}

// ---------------------------------------------------------------------------
// TDM: DMA a 128 x KSLAB bf16 tile (row-major, row stride = rowElems) from
// global into LDS at ldsAddr, inserting 16B of padding after every row so the
// padded LDS row stride (KSLAB+8 elements) is bank-conflict-free for the
// B-fragment ds_load_b128 pattern. D# layout per CDNA5 ISA 8.3/8.4.
// Issued by one wave; completion via s_wait_tensorcnt.
// ---------------------------------------------------------------------------
template <int KSLAB>
__device__ __forceinline__ void tdm_stage_w(const __bf16* gsrc, int rowElems,
                                            unsigned ldsAddr) {
  unsigned long long ga = (unsigned long long)(uintptr_t)gsrc;
  u32x4 g0;
  g0[0] = 1u;                                  // count=1, user mode
  g0[1] = ldsAddr;                             // lds_addr (bytes)
  g0[2] = (unsigned)ga;                        // global_addr[31:0]
  g0[3] = (unsigned)((ga >> 32) & 0x01FFFFFFu) | (2u << 30);  // addr[56:32], type=2
  constexpr unsigned padItv = (KSLAB == 128) ? 5u : 4u;  // 64 / 32 dwords per row
  i32x8 g1;
  g1[0] = (int)((1u << 16)          // data_size: 2 bytes
              | (1u << 20)          // pad_enable
              | (padItv << 22)      // pad_interval (one row)
              | (3u << 25));        // pad_amount: 4 dwords = 16B
  g1[1] = (int)(((unsigned)rowElems & 0xFFFFu) << 16);                 // tensor_dim0 lo
  g1[2] = (int)((((unsigned)rowElems >> 16) & 0xFFFFu) | (128u << 16));// dim0 hi, dim1=128
  g1[3] = (int)((unsigned)KSLAB << 16);        // tensor_dim1 hi=0, tile_dim0=KSLAB
  g1[4] = (int)128;                            // tile_dim1=128, tile_dim2=0
  g1[5] = (int)rowElems;                       // tensor_dim0_stride[31:0]
  g1[6] = 0;
  g1[7] = 0;
  i32x4 z4 = (i32x4){0, 0, 0, 0};
#if __clang_major__ >= 23
  i32x8 z8 = (i32x8){0, 0, 0, 0, 0, 0, 0, 0};
  __builtin_amdgcn_tensor_load_to_lds(g0, g1, z4, z4, z8, 0);
#else
  __builtin_amdgcn_tensor_load_to_lds(g0, g1, z4, z4, 0);
#endif
  __builtin_amdgcn_s_wait_tensorcnt(0);
}

// ---------------------------------------------------------------------------
// GEMM: C[:, 0:128 @ stride CS] = A[M,K] (bf16) x W[128,K]^T (bf16), f32 acc.
// W is staged into LDS by the Tensor Data Mover once per block (per K-slab),
// with 16B/row padding for conflict-free ds_load_b128 B fragments at
// compile-time immediate offsets. A fragments live in registers. B is
// double-buffered at tile granularity so each tile's ds_load clause is issued
// a full 2-4 WMMA chain before its s_wait_dscnt.
// Fragment k order is contiguous, k = 16*(lane/16) + s, same for A and B (16-bit A 16x32 / B 32x16, wave32).
// Requires M % 16 == 0 (true: N=50000, E=800000).
// ---------------------------------------------------------------------------
template <int K, int CS>
__global__ __launch_bounds__(256) void gemm_wmma_bf16(
    const __bf16* __restrict__ A, const __bf16* __restrict__ W,
    float* __restrict__ C, int tiles_m) {
  constexpr int KS    = K / 32;                // total k-steps
  constexpr int KSLAB = (K < 128) ? K : 128;   // columns staged per TDM op
  constexpr int NSLAB = K / KSLAB;
  constexpr int SKS   = KSLAB / 32;            // k-steps per slab
  constexpr int SROW  = KSLAB + 8;             // padded LDS row stride (elements)
  __shared__ __bf16 sW[128 * SROW];            // <= 34 KB

  int waveId = (int)((blockIdx.x * blockDim.x + threadIdx.x) >> 5);
  bool active = waveId < tiles_m;
  int wid = active ? waveId : (tiles_m - 1);   // clamp; all waves hit barriers
  int lane = (int)(threadIdx.x & 31);
  int r = lane & 15;
  bool hi = lane >= 16;

  // Preload the entire A row-slab (<= 64 VGPRs for K=256).
  const __bf16* Ar = A + (size_t)(wid * 16 + r) * K + (hi ? 16 : 0);
  v16bf a[KS];
#pragma unroll
  for (int ks = 0; ks < KS; ++ks) {
    v8bf alo = *(const v8bf*)(Ar + ks * 32);
    v8bf ahi = *(const v8bf*)(Ar + ks * 32 + 8);
    a[ks] = __builtin_shufflevector(alo, ahi,
        0, 1, 2, 3, 4, 5, 6, 7, 8, 9, 10, 11, 12, 13, 14, 15);
  }

  const __bf16* sB = sW + r * SROW + (hi ? 16 : 0);
  v8f acc[8];
#pragma unroll
  for (int t = 0; t < 8; ++t) acc[t] = v8f{};

#pragma unroll
  for (int sl = 0; sl < NSLAB; ++sl) {
    if (sl) __syncthreads();                   // LDS reuse fence
    if (threadIdx.x < 32)                      // wave 0 drives the TDM
      tdm_stage_w<KSLAB>(W + (size_t)sl * KSLAB, K,
                         (unsigned)(uintptr_t)&sW[0]);
    __syncthreads();

    v16bf b[2][SKS];
#pragma unroll
    for (int ks = 0; ks < SKS; ++ks)           // prime buffer 0 (tile 0)
      b[0][ks] = *(const v16bf*)(sB + ks * 32);
#pragma unroll
    for (int t = 0; t < 8; ++t) {
      if (t + 1 < 8) {                         // issue next tile's loads first
#pragma unroll
        for (int ks = 0; ks < SKS; ++ks)
          b[(t + 1) & 1][ks] =
              *(const v16bf*)(sB + (t + 1) * 16 * SROW + ks * 32);
      }
#pragma unroll
      for (int ks = 0; ks < SKS; ++ks)
        acc[t] = __builtin_amdgcn_wmma_f32_16x16x32_bf16(
            false, a[sl * SKS + ks], false, b[t & 1][ks], (short)0, acc[t],
            false, false);
    }
  }

  if (active) {
    float* Cp = C + (size_t)(waveId * 16 + (hi ? 8 : 0)) * CS + r;
#pragma unroll
    for (int t = 0; t < 8; ++t)
#pragma unroll
      for (int rr = 0; rr < 8; ++rr)
        Cp[(size_t)rr * CS + t * 16] = acc[t][rr];
  }
}

// ---------------------------------------------------------------------------
// Edge fuse: one wave per edge. f[e] = leaky(f_ni[src]+f_nj[dst]+f_fij[e]+bias)
// in-place over f_fij; per-head logits = sum_{d<32} f*attn (8 lanes per head).
// ---------------------------------------------------------------------------
__global__ __launch_bounds__(256) void edge_fuse(
    const float* __restrict__ fni, const float* __restrict__ fnj,
    float* __restrict__ f, const float* __restrict__ bias,
    const float* __restrict__ attn, const int* __restrict__ src,
    const int* __restrict__ dst, float* __restrict__ logits, int E) {
  int e = (int)((blockIdx.x * blockDim.x + threadIdx.x) >> 5);
  if (e >= E) return;
  int lane = (int)(threadIdx.x & 31);
  int c = lane * 4;
  int s = src[e], d = dst[e];
  const float4 va    = *(const float4*)(fni  + (size_t)s * 128 + c);
  const float4 vb    = *(const float4*)(fnj  + (size_t)d * 128 + c);
  const float4 vf    = *(const float4*)(f    + (size_t)e * 128 + c);
  const float4 vbias = *(const float4*)(bias + c);
  const float4 vat   = *(const float4*)(attn + c);
  float x0 = va.x + vb.x + vf.x + vbias.x; x0 = x0 >= 0.f ? x0 : 0.2f * x0;
  float x1 = va.y + vb.y + vf.y + vbias.y; x1 = x1 >= 0.f ? x1 : 0.2f * x1;
  float x2 = va.z + vb.z + vf.z + vbias.z; x2 = x2 >= 0.f ? x2 : 0.2f * x2;
  float x3 = va.w + vb.w + vf.w + vbias.w; x3 = x3 >= 0.f ? x3 : 0.2f * x3;
  *(float4*)(f + (size_t)e * 128 + c) = make_float4(x0, x1, x2, x3);
  float lg = x0 * vat.x + x1 * vat.y + x2 * vat.z + x3 * vat.w;
  lg += __shfl_xor(lg, 1);
  lg += __shfl_xor(lg, 2);
  lg += __shfl_xor(lg, 4);
  if ((lane & 7) == 0) logits[(size_t)e * 4 + (lane >> 3)] = lg;
}

// Segment max over dst via float atomic-max (int/uint trick; m init = -inf).
__global__ __launch_bounds__(256) void seg_max(
    const float* __restrict__ logits, const int* __restrict__ dst,
    float* __restrict__ m, int n) {
  int i = (int)(blockIdx.x * blockDim.x + threadIdx.x);
  if (i >= n) return;
  int e = i >> 2, h = i & 3;
  float v = logits[i];
  float* addr = m + (size_t)dst[e] * 4 + h;
  if (v >= 0.f) atomicMax((int*)addr, __float_as_int(v));
  else          atomicMin((unsigned int*)addr, __float_as_uint(v));
}

// ex = exp(logit - m[dst]); logits overwritten with ex; s[dst] += ex.
__global__ __launch_bounds__(256) void exp_sum(
    float* __restrict__ logits, const int* __restrict__ dst,
    const float* __restrict__ m, float* __restrict__ ssum, int n) {
  int i = (int)(blockIdx.x * blockDim.x + threadIdx.x);
  if (i >= n) return;
  int e = i >> 2, h = i & 3;
  int d = dst[e];
  float ex = __expf(logits[i] - m[(size_t)d * 4 + h]);
  logits[i] = ex;
  atomicAdd(ssum + (size_t)d * 4 + h, ex);
}

// h_out[dst] += h_src[src] * (ex/s[dst]). One wave per edge, 8 ch/lane,
// head = lane>>3 (64 channels per head). Atomics land in the 192MB L2.
__global__ __launch_bounds__(256) void scatter_msg(
    const float* __restrict__ hsrc, const float* __restrict__ ex,
    const float* __restrict__ ssum, const int* __restrict__ src,
    const int* __restrict__ dst, float* __restrict__ hout, int E) {
  int e = (int)((blockIdx.x * blockDim.x + threadIdx.x) >> 5);
  if (e >= E) return;
  int lane = (int)(threadIdx.x & 31);
  int head = lane >> 3;
  int s = src[e], d = dst[e];
  float a = ex[(size_t)e * 4 + head] / ssum[(size_t)d * 4 + head];
  int c = lane * 8;
  const float* hs = hsrc + (size_t)s * 256 + c;
  float* ho = hout + (size_t)d * 256 + c;
#pragma unroll
  for (int i = 0; i < 8; ++i) atomicAdd(ho + i, hs[i] * a);
}

__global__ __launch_bounds__(256) void fill_f32(float* __restrict__ p, float v, size_t n) {
  size_t i = (size_t)blockIdx.x * blockDim.x + threadIdx.x;
  if (i < n) p[i] = v;
}

// ---------------------------------------------------------------------------
extern "C" void kernel_launch(void* const* d_in, const int* in_sizes, int n_in,
                              void* d_out, int out_size, void* d_ws, size_t ws_size,
                              hipStream_t stream) {
  (void)in_sizes; (void)n_in; (void)out_size; (void)ws_size;
  const int* src = (const int*)d_in[0];
  const int* dst = (const int*)d_in[1];
  const float* nfeats  = (const float*)d_in[2];
  const float* efeats  = (const float*)d_in[3];
  const float* l0_attn = (const float*)d_in[8];
  const float* l0_bias = (const float*)d_in[9];
  const float* l1_attn = (const float*)d_in[14];
  const float* l1_bias = (const float*)d_in[15];
  const float* prelu_a = (const float*)d_in[16];

  const int N = NN, E = EE;
  float* out   = (float*)d_out;
  float* out_h = out;                       // [N,256] final node features
  float* out_f = out + (size_t)N * 256;     // [E,128] final edge features

  // ---- workspace carve-up -------------------------------------------------
  float* ws    = (float*)d_ws;
  float* f_ni  = ws;                            // N*128 f32
  float* f_nj  = f_ni  + (size_t)N * 128;       // N*128
  float* h_src = f_nj  + (size_t)N * 128;       // N*256
  float* f0    = h_src + (size_t)N * 256;       // E*128 (layer-0 f, in-place)
  float* lgts  = f0    + (size_t)E * 128;       // E*4 (logits -> ex)
  float* mbuf  = lgts  + (size_t)E * 4;         // N*4
  float* sbuf  = mbuf  + (size_t)N * 4;         // N*4
  float* h0    = sbuf  + (size_t)N * 4;         // N*256 (layer-0 h)
  __bf16* bws  = (__bf16*)(h0 + (size_t)N * 256);
  __bf16* nf_bf = bws;                          // N*256 bf16
  __bf16* ef_bf = nf_bf + (size_t)N * 256;      // E*64
  __bf16* h0_bf = ef_bf + (size_t)E * 64;       // N*256
  __bf16* f0_bf = h0_bf + (size_t)N * 256;      // E*128
  __bf16* wb    = f0_bf + (size_t)E * 128;      // weights (bf16)
  __bf16* w0ni   = wb;
  __bf16* w0nj   = w0ni   + 128 * 256;
  __bf16* w0fij  = w0nj   + 128 * 256;
  __bf16* w0node = w0fij  + 128 * 64;
  __bf16* w1ni   = w0node + 256 * 256;
  __bf16* w1nj   = w1ni   + 128 * 256;
  __bf16* w1fij  = w1nj   + 128 * 256;
  __bf16* w1node = w1fij  + 128 * 128;

  auto cvt = [&](const float* in, __bf16* o, size_t n) {
    size_t n4 = n / 4;
    cvt_bf16<<<(int)((n4 + 255) / 256), 256, 0, stream>>>(
        (const float4*)in, (uint2*)o, n4);
  };
  auto fill = [&](float* p, float v, size_t n) {
    fill_f32<<<(int)((n + 255) / 256), 256, 0, stream>>>(p, v, n);
  };

  const int tN = N / 16, tE = E / 16;           // exact (both % 16 == 0)
  const int gN = (tN + 7) / 8, gE = (tE + 7) / 8;
  const int eWaveBlocks = (E * 32 + 255) / 256; // one wave per edge
  const int ehBlocks    = (E * 4 + 255) / 256;  // one thread per (edge,head)
  const float NEG_INF = -__builtin_huge_valf();

  // ---- one-time bf16 conversions ------------------------------------------
  cvt(nfeats, nf_bf, (size_t)N * 256);
  cvt(efeats, ef_bf, (size_t)E * 64);
  cvt((const float*)d_in[4],  w0ni,   128 * 256);
  cvt((const float*)d_in[5],  w0nj,   128 * 256);
  cvt((const float*)d_in[6],  w0fij,  128 * 64);
  cvt((const float*)d_in[7],  w0node, 256 * 256);
  cvt((const float*)d_in[10], w1ni,   128 * 256);
  cvt((const float*)d_in[11], w1nj,   128 * 256);
  cvt((const float*)d_in[12], w1fij,  128 * 128);
  cvt((const float*)d_in[13], w1node, 256 * 256);

  // ---- Layer 0: nfeats [N,256], efeats [E,64] -> h0 [N,256], f0 [E,128] ----
  gemm_wmma_bf16<256, 128><<<gN, 256, 0, stream>>>(nf_bf, w0ni, f_ni, tN);
  gemm_wmma_bf16<256, 128><<<gN, 256, 0, stream>>>(nf_bf, w0nj, f_nj, tN);
  gemm_wmma_bf16< 64, 128><<<gE, 256, 0, stream>>>(ef_bf, w0fij, f0, tE);
  gemm_wmma_bf16<256, 256><<<gN, 256, 0, stream>>>(nf_bf, w0node, h_src, tN);
  gemm_wmma_bf16<256, 256><<<gN, 256, 0, stream>>>(nf_bf, w0node + 128 * 256,
                                                   h_src + 128, tN);
  fill(mbuf, NEG_INF, (size_t)N * 4);
  fill(sbuf, 0.f,     (size_t)N * 4);
  fill(h0,   0.f,     (size_t)N * 256);
  edge_fuse<<<eWaveBlocks, 256, 0, stream>>>(f_ni, f_nj, f0, l0_bias, l0_attn,
                                             src, dst, lgts, E);
  seg_max<<<ehBlocks, 256, 0, stream>>>(lgts, dst, mbuf, E * 4);
  exp_sum<<<ehBlocks, 256, 0, stream>>>(lgts, dst, mbuf, sbuf, E * 4);
  scatter_msg<<<eWaveBlocks, 256, 0, stream>>>(h_src, lgts, sbuf, src, dst, h0, E);
  // shared PReLU fused with bf16 conversion (outputs feed only layer-1 GEMMs)
  {
    size_t n4 = (size_t)N * 256 / 4;
    prelu_cvt_bf16<<<(int)((n4 + 255) / 256), 256, 0, stream>>>(
        (const float4*)h0, prelu_a, (uint2*)h0_bf, n4);
    n4 = (size_t)E * 128 / 4;
    prelu_cvt_bf16<<<(int)((n4 + 255) / 256), 256, 0, stream>>>(
        (const float4*)f0, prelu_a, (uint2*)f0_bf, n4);
  }

  // ---- Layer 1: h0 [N,256], f0 [E,128] -> out_h [N,256], out_f [E,128] ----
  gemm_wmma_bf16<256, 128><<<gN, 256, 0, stream>>>(h0_bf, w1ni, f_ni, tN);
  gemm_wmma_bf16<256, 128><<<gN, 256, 0, stream>>>(h0_bf, w1nj, f_nj, tN);
  gemm_wmma_bf16<128, 128><<<gE, 256, 0, stream>>>(f0_bf, w1fij, out_f, tE);
  gemm_wmma_bf16<256, 256><<<gN, 256, 0, stream>>>(h0_bf, w1node, h_src, tN);
  gemm_wmma_bf16<256, 256><<<gN, 256, 0, stream>>>(h0_bf, w1node + 128 * 256,
                                                   h_src + 128, tN);
  fill(mbuf, NEG_INF, (size_t)N * 4);
  fill(sbuf, 0.f,     (size_t)N * 4);
  fill(out_h, 0.f,    (size_t)N * 256);
  edge_fuse<<<eWaveBlocks, 256, 0, stream>>>(f_ni, f_nj, out_f, l1_bias, l1_attn,
                                             src, dst, lgts, E);
  seg_max<<<ehBlocks, 256, 0, stream>>>(lgts, dst, mbuf, E * 4);
  exp_sum<<<ehBlocks, 256, 0, stream>>>(lgts, dst, mbuf, sbuf, E * 4);
  scatter_msg<<<eWaveBlocks, 256, 0, stream>>>(h_src, lgts, sbuf, src, dst, out_h, E);
  // encoding=False: no activation on final layer.
}
